// HPLFlowNet_68418829025450
// MI455X (gfx1250) — matrix operand order repaired and checked by reading, not compiled, not run on hardware
//
#include <hip/hip_runtime.h>

typedef __attribute__((ext_vector_type(16))) _Float16 v16h;
typedef __attribute__((ext_vector_type(8)))  _Float16 v8h;
typedef __attribute__((ext_vector_type(4)))  _Float16 v4h;
typedef __attribute__((ext_vector_type(8)))  float    v8f;

__device__ __forceinline__ float lrelu(float x){ return x > 0.f ? x : 0.1f * x; }

__device__ __forceinline__ v16h mk16(v8h lo, v8h hi){
  v16h r;
#pragma unroll
  for (int i = 0; i < 8; ++i){ r[i] = lo[i]; r[i + 8] = hi[i]; }
  return r;
}

// A-fragment (16x32 f16): row-major W (Mp x Kp). Lane l -> row mbase+(l&15).
// lanes 0-15 hold K = ks+{0..15}; lanes 16-31 hold K = ks+{16..31}.
__device__ __forceinline__ v16h load_afrag(const _Float16* __restrict__ W, int Kp,
                                           int mbase, int ks, int lane){
  const _Float16* r = W + (size_t)(mbase + (lane & 15)) * Kp + ks + ((lane & 16) ? 16 : 0);
  v8h lo = *(const v8h*)(r);
  v8h hi = *(const v8h*)(r + 8);
  return mk16(lo, hi);
}

// B-fragment (32x16 f16) from site-major X. Lane l -> site sbase+(l&15),
// K = ks + ((l&16)?16:0) + 0..15 contiguous.
__device__ __forceinline__ v16h load_bfrag(const _Float16* __restrict__ X, size_t ldx,
                                           int sbase, int ks, int lane){
  const _Float16* r = X + (size_t)(sbase + (lane & 15)) * ldx + ks + ((lane & 16) ? 16 : 0);
  v8h lo = *(const v8h*)(r);
  v8h hi = *(const v8h*)(r + 8);
  return mk16(lo, hi);
}

// C/D layout: lane l, VGPR v -> (M = v + ((l&16)?8:0), N = l&15)
__device__ __forceinline__ void store_cfrag_f16(_Float16* dstRow, int mbase, v8f c,
                                                const float* __restrict__ bias,
                                                int lane, int act){
  int moff = mbase + ((lane & 16) ? 8 : 0);
  v8h o;
#pragma unroll
  for (int v = 0; v < 8; ++v){
    float x = c[v] + bias[moff + v];
    o[v] = (_Float16)(act ? lrelu(x) : x);
  }
  *(v8h*)(dstRow + moff) = o;
}

// ---------------------------------------------------------------- weight repack
__global__ void k_pack_dense(const float* __restrict__ src, _Float16* __restrict__ dst, int n){
  int i = blockIdx.x * 256 + threadIdx.x;
  if (i < n) dst[i] = (_Float16)src[i];
}

// src (O, Isrc, NN) f32 -> dst (O, Kp) f16 with K = n*Ipad + i; zero pad i>=Isrc, n>=NN
__global__ void k_pack_blur(const float* __restrict__ src, _Float16* __restrict__ dst,
                            int O, int Isrc, int Ipad, int NNn, int Kp){
  int t = blockIdx.x * 256 + threadIdx.x;
  if (t >= O * Kp) return;
  int o = t / Kp, k = t % Kp;
  int n = k / Ipad, i = k % Ipad;
  float v = (n < NNn && i < Isrc) ? src[((size_t)o * Isrc + i) * NNn + n] : 0.f;
  dst[t] = (_Float16)v;
}

// ---------------------------------------------------------------- per-point MLP
__global__ __launch_bounds__(256) void k_mlp3(
    const float* __restrict__ pc, const float* __restrict__ w1, const float* __restrict__ b1,
    const float* __restrict__ w2, const float* __restrict__ b2,
    const float* __restrict__ w3, const float* __restrict__ b3,
    float* __restrict__ feat, int N){
  __shared__ float s1[96], sb1[32], s2[1024], sb2[32], s3[2048], sb3[64];
  int tid = threadIdx.x;
  for (int i = tid; i < 96;   i += 256) s1[i] = w1[i];
  for (int i = tid; i < 1024; i += 256) s2[i] = w2[i];
  for (int i = tid; i < 2048; i += 256) s3[i] = w3[i];
  if (tid < 32){ sb1[tid] = b1[tid]; sb2[tid] = b2[tid]; }
  if (tid < 64)  sb3[tid] = b3[tid];
  __syncthreads();
  int n = blockIdx.x * 256 + tid;
  if (n >= N) return;
  float x0 = pc[n], x1 = pc[N + n], x2 = pc[2 * N + n];
  float h1[32], h2[32];
#pragma unroll 4
  for (int o = 0; o < 32; ++o)
    h1[o] = lrelu(s1[o*3]*x0 + s1[o*3+1]*x1 + s1[o*3+2]*x2 + sb1[o]);
#pragma unroll 2
  for (int o = 0; o < 32; ++o){
    float s = sb2[o];
    for (int i = 0; i < 32; ++i) s += s2[o*32+i]*h1[i];
    h2[o] = lrelu(s);
  }
#pragma unroll 2
  for (int o = 0; o < 64; ++o){
    float s = sb3[o];
    for (int i = 0; i < 32; ++i) s += s3[o*32+i]*h2[i];
    feat[(size_t)n*64 + o] = lrelu(s);
  }
}

// ---------------------------------------------------------------- splat (scatter-add)
__global__ void k_splat(const float* __restrict__ el, const float* __restrict__ feat,
                        const float* __restrict__ bary, const int* __restrict__ off,
                        float* __restrict__ lat, int N){
  int t = blockIdx.x * 256 + threadIdx.x;
  int n = t >> 2, cb = (t & 3) * 17;     // 68 channels -> 4 groups of 17
  if (n >= N) return;
  float vals[17];
#pragma unroll
  for (int j = 0; j < 17; ++j){
    int c = cb + j;
    vals[j] = (c < 4) ? el[(size_t)c * N + n] : feat[(size_t)n * 64 + (c - 4)];
  }
  for (int v = 0; v < 5; ++v){
    float w = bary[(size_t)v * N + n];
    int idx = off[(size_t)v * N + n];
    float* dst = lat + (size_t)idx * 68 + cb;
#pragma unroll
    for (int j = 0; j < 17; ++j) atomicAdd(dst + j, w * vals[j]);
  }
}

// ---------------------------------------------------------------- blur with sw1
// K layout: n*72 + i (each neighbor block padded 68->72), Kp = 1088 (34 wmma steps)
// 32 sites/block: LDS 70 KB (CDNA5 allows up to 320 KB per workgroup)
#define BKS 1096   // 1088 + 8 f16 pad: site stride = 548 dw, 548%64=36 -> no bank clash
__global__ __launch_bounds__(256) void k_blur_sw1(
    const float* __restrict__ lat,          // (H,68) f32
    const int* __restrict__ nb,             // (15,H)
    const _Float16* __restrict__ W,         // (64,1088) packed
    const float* __restrict__ bias,         // 64
    _Float16* __restrict__ Y,               // (H,64)
    int H){
  __shared__ _Float16 Bs[32 * BKS];
  int tid = threadIdx.x;
  int hbase = blockIdx.x * 32;
  // zero the pad slots: per site 15*4 (i=68..71) + 8 (K 1080..1087) = 68
  for (int t = tid; t < 32 * 68; t += 256){
    int site = t / 68, j = t % 68;
    int idx = (j < 60) ? ((j >> 2) * 72 + 68 + (j & 3)) : (1080 + (j - 60));
    Bs[site * BKS + idx] = (_Float16)0.f;
  }
  for (int p = tid; p < 480; p += 256){     // 32 sites x 15 neighbors
    int site = p / 15, n = p % 15;
    int g = nb[(size_t)n * H + hbase + site];
    const float4* src = (const float4*)(lat + (size_t)g * 68);
    _Float16* dst = &Bs[site * BKS + n * 72];
#pragma unroll
    for (int q = 0; q < 17; ++q){
      float4 f = src[q];
      v4h h; h[0] = (_Float16)f.x; h[1] = (_Float16)f.y;
             h[2] = (_Float16)f.z; h[3] = (_Float16)f.w;
      *(v4h*)(dst + 4 * q) = h;
    }
  }
  __syncthreads();
  int wid = tid >> 5, lane = tid & 31;      // 8 waves = 4 M-tiles x 2 N-tiles
  int mt = wid & 3, nt = wid >> 2;
  int siteL = nt * 16 + (lane & 15);
  int koff = (lane & 16) ? 16 : 0;
  v8f acc = {};
  for (int ks = 0; ks < 1088; ks += 32){
    v16h a = load_afrag(W, 1088, mt * 16, ks, lane);
    const v8h* bp = (const v8h*)&Bs[siteL * BKS + ks + koff];
    v16h b = mk16(bp[0], bp[1]);
    acc = __builtin_amdgcn_wmma_f32_16x16x32_f16(false, a, false, b, (short)0, acc, false, false);
  }
  store_cfrag_f16(Y + (size_t)(hbase + siteL) * 64, mt * 16, acc, bias, lane, 1);
}

// ---------------------------------------------------------------- p-blur (async-to-LDS gather)
// 32 sites/block: LDS B panel 123 KB; 8 waves = 4 M-pairs x 2 N-tiles,
// each wave computes 2 M-tiles reusing one B fragment.
#define PKS 1928   // 1920 + 8 f16 pad
__global__ __launch_bounds__(256) void k_pblur(
    const _Float16* __restrict__ cat,       // (H,128) f16
    const int* __restrict__ nb,             // (15,H)
    const _Float16* __restrict__ W,         // (128,1920) packed
    const float* __restrict__ bias,         // 128
    _Float16* __restrict__ Y,               // (H,128)
    int H){
  __shared__ _Float16 Bs[32 * PKS];
  int tid = threadIdx.x;
  int hbase = blockIdx.x * 32;
  for (int t = tid; t < 960; t += 256){     // 480 pairs x 2 half-rows of 128B
    int pair = t >> 1, half = t & 1;
    int site = pair / 15, n = pair % 15;
    int g = nb[(size_t)n * H + hbase + site];
    const _Float16* src = cat + (size_t)g * 128 + half * 64;
    unsigned lds = (unsigned)((site * PKS + n * 128 + half * 64) * 2);
#pragma unroll
    for (int q = 0; q < 8; ++q){
      asm volatile("global_load_async_to_lds_b128 %0, %1, off offset:%2"
                   :: "v"(lds), "v"(src), "n"(q * 16) : "memory");
    }
  }
  asm volatile("s_wait_asynccnt 0x0" ::: "memory");
  __syncthreads();
  int wid = tid >> 5, lane = tid & 31;
  int mt0 = (wid & 3) * 2;                  // M-pair
  int nt  = wid >> 2;                       // N-tile
  int siteL = nt * 16 + (lane & 15);
  int koff = (lane & 16) ? 16 : 0;
  const _Float16* arow = W + (size_t)(mt0 * 16 + (lane & 15)) * 1920;
  v8f acc0 = {}, acc1 = {};
  for (int ks = 0; ks < 1920; ks += 32){
    if (ks + 256 < 1920) __builtin_prefetch(arow + ks + 256, 0, 0);
    v16h a0 = load_afrag(W, 1920, mt0 * 16, ks, lane);
    v16h a1 = load_afrag(W, 1920, mt0 * 16 + 16, ks, lane);
    const v8h* bp = (const v8h*)&Bs[siteL * PKS + ks + koff];
    v16h b = mk16(bp[0], bp[1]);
    acc0 = __builtin_amdgcn_wmma_f32_16x16x32_f16(false, a0, false, b, (short)0, acc0, false, false);
    acc1 = __builtin_amdgcn_wmma_f32_16x16x32_f16(false, a1, false, b, (short)0, acc1, false, false);
  }
  _Float16* dstRow = Y + (size_t)(hbase + siteL) * 128;
  store_cfrag_f16(dstRow, mt0 * 16,      acc0, bias, lane, 1);
  store_cfrag_f16(dstRow, mt0 * 16 + 16, acc1, bias, lane, 1);
}

// ---------------------------------------------------------------- generic dense GEMM: Y = act(W @ X^T + b)
// Each wave: one M-tile x two site-tiles (A fragment reused twice).
__global__ __launch_bounds__(256) void k_gemm(
    const _Float16* __restrict__ X, int ldx,
    const _Float16* __restrict__ W, int Kp, int Mp,
    const float* __restrict__ bias,
    _Float16* __restrict__ Y, int ldy, int colOff,
    int NS, int act){
  int gw = (blockIdx.x * 256 + threadIdx.x) >> 5;
  int lane = threadIdx.x & 31;
  int mtn = Mp >> 4;
  int mt = gw % mtn;
  int sb0 = (gw / mtn) * 32;
  if (sb0 >= NS) return;
  v8f acc0 = {}, acc1 = {};
  for (int ks = 0; ks < Kp; ks += 32){
    v16h a  = load_afrag(W, Kp, mt * 16, ks, lane);
    v16h b0 = load_bfrag(X, (size_t)ldx, sb0,      ks, lane);
    v16h b1 = load_bfrag(X, (size_t)ldx, sb0 + 16, ks, lane);
    acc0 = __builtin_amdgcn_wmma_f32_16x16x32_f16(false, a, false, b0, (short)0, acc0, false, false);
    acc1 = __builtin_amdgcn_wmma_f32_16x16x32_f16(false, a, false, b1, (short)0, acc1, false, false);
  }
  int s0 = sb0 + (lane & 15);
  store_cfrag_f16(Y + (size_t)s0 * ldy + colOff,        mt * 16, acc0, bias, lane, act);
  store_cfrag_f16(Y + (size_t)(s0 + 16) * ldy + colOff, mt * 16, acc1, bias, lane, act);
}

// ---------------------------------------------------------------- correlation: prod -> cw1 -> cw2 -> mean_k
__global__ __launch_bounds__(64) void k_corr(
    const _Float16* __restrict__ lat1,   // (H,128), cols 0..63 valid
    const _Float16* __restrict__ lat2,   // (H,64)
    const int* __restrict__ cidx,        // (32,H)
    const _Float16* __restrict__ Wc1,    // (32,64)
    const float* __restrict__ cb1,
    const _Float16* __restrict__ Wc2,    // (32,32)
    const float* __restrict__ cb2,
    _Float16* __restrict__ agg,          // (H,32)
    int H){
  __shared__ _Float16 Bp[16 * 72];
  __shared__ _Float16 Bm[16 * 40];
  __shared__ int gsi[16];
  int tid = threadIdx.x, lane = tid & 31, wid = tid >> 5;
  int hbase = blockIdx.x * 16;
  int siteL = lane & 15;
  int koff = (lane & 16) ? 16 : 0;
  int moff = wid * 16 + ((lane & 16) ? 8 : 0);
  v8f aggacc = {};
  for (int k = 0; k < 32; ++k){
    if (tid < 16) gsi[tid] = cidx[(size_t)k * H + hbase + tid];
    __syncthreads();
    for (int e = tid; e < 1024; e += 64){      // elementwise lat1*g2 panel
      int site = e >> 6, c = e & 63;
      float a = (float)lat1[(size_t)(hbase + site) * 128 + c];
      float b = (float)lat2[(size_t)gsi[site] * 64 + c];
      Bp[site * 72 + c] = (_Float16)(a * b);
    }
    __syncthreads();
    v8f c1 = {};
#pragma unroll
    for (int ks = 0; ks < 64; ks += 32){
      v16h a = load_afrag(Wc1, 64, wid * 16, ks, lane);
      const v8h* bp = (const v8h*)&Bp[siteL * 72 + ks + koff];
      v16h b = mk16(bp[0], bp[1]);
      c1 = __builtin_amdgcn_wmma_f32_16x16x32_f16(false, a, false, b, (short)0, c1, false, false);
    }
    v8h mh;
#pragma unroll
    for (int v = 0; v < 8; ++v) mh[v] = (_Float16)lrelu(c1[v] + cb1[moff + v]);
    *(v8h*)&Bm[siteL * 40 + moff] = mh;        // transpose m1 through LDS
    __syncthreads();
    v8f d = {};
    {
      v16h a2 = load_afrag(Wc2, 32, wid * 16, 0, lane);
      const v8h* bp = (const v8h*)&Bm[siteL * 40 + koff];
      v16h b = mk16(bp[0], bp[1]);
      d = __builtin_amdgcn_wmma_f32_16x16x32_f16(false, a2, false, b, (short)0, d, false, false);
    }
#pragma unroll
    for (int v = 0; v < 8; ++v) aggacc[v] += lrelu(d[v] + cb2[moff + v]);
  }
  v8h o;
#pragma unroll
  for (int v = 0; v < 8; ++v) o[v] = (_Float16)(aggacc[v] * (1.f / 32.f));
  *(v8h*)&agg[(size_t)(hbase + siteL) * 32 + moff] = o;
}

// ---------------------------------------------------------------- slice
__global__ void k_slice(const _Float16* __restrict__ P, const float* __restrict__ bary,
                        const int* __restrict__ off, _Float16* __restrict__ pts, int N){
  int t = blockIdx.x * 256 + threadIdx.x;
  int n = t >> 2, ch = (t & 3) * 32;
  if (n >= N) return;
  float acc[32] = {};
  for (int v = 0; v < 5; ++v){
    float w = bary[(size_t)v * N + n];
    const _Float16* src = P + (size_t)off[(size_t)v * N + n] * 128 + ch;
#pragma unroll
    for (int j = 0; j < 32; ++j) acc[j] += w * (float)src[j];
  }
#pragma unroll
  for (int j = 0; j < 32; ++j) pts[(size_t)n * 128 + ch + j] = (_Float16)acc[j];
}

// ---------------------------------------------------------------- final fw4 (3x64)
__global__ void k_final(const _Float16* __restrict__ h, const float* __restrict__ w4,
                        const float* __restrict__ b4, float* __restrict__ out, int N){
  int n = blockIdx.x * 256 + threadIdx.x;
  if (n >= N) return;
  float s0 = b4[0], s1 = b4[1], s2 = b4[2];
  for (int i = 0; i < 64; ++i){
    float x = (float)h[(size_t)n * 64 + i];
    s0 += w4[i] * x; s1 += w4[64 + i] * x; s2 += w4[128 + i] * x;
  }
  out[n] = s0; out[(size_t)N + n] = s1; out[(size_t)2 * N + n] = s2;
}

// ================================================================ host
extern "C" void kernel_launch(void* const* d_in, const int* in_sizes, int n_in,
                              void* d_out, int out_size, void* d_ws, size_t ws_size,
                              hipStream_t stream){
  const float* pc1   = (const float*)d_in[0];
  const float* pc2   = (const float*)d_in[1];
  const float* el1   = (const float*)d_in[2];
  const float* el2   = (const float*)d_in[3];
  const float* bary1 = (const float*)d_in[4];
  const float* bary2 = (const float*)d_in[5];
  const int*   off1  = (const int*)d_in[6];
  const int*   off2  = (const int*)d_in[7];
  const int*   nb1   = (const int*)d_in[8];
  const int*   nb2   = (const int*)d_in[9];
  const int*   cidx  = (const int*)d_in[10];
  const float *w1 = (const float*)d_in[11], *b1 = (const float*)d_in[12];
  const float *w2 = (const float*)d_in[13], *b2 = (const float*)d_in[14];
  const float *w3 = (const float*)d_in[15], *b3 = (const float*)d_in[16];
  const float *sw1 = (const float*)d_in[17], *sb1 = (const float*)d_in[18];
  const float *sw2 = (const float*)d_in[19], *sb2 = (const float*)d_in[20];
  const float *cw1 = (const float*)d_in[21], *cb1 = (const float*)d_in[22];
  const float *cw2 = (const float*)d_in[23], *cb2 = (const float*)d_in[24];
  const float *dw1 = (const float*)d_in[25], *db1 = (const float*)d_in[26];
  const float *dw2 = (const float*)d_in[27], *db2 = (const float*)d_in[28];
  const float *pw1 = (const float*)d_in[29], *pb1 = (const float*)d_in[30];
  const float *pw2 = (const float*)d_in[31], *pb2 = (const float*)d_in[32];
  const float *fw2 = (const float*)d_in[33], *fb2 = (const float*)d_in[34];
  const float *fw3 = (const float*)d_in[35], *fb3 = (const float*)d_in[36];
  const float *fw4 = (const float*)d_in[37], *fb4 = (const float*)d_in[38];

  const int N = in_sizes[0] / 3;
  const int H = in_sizes[8] / 15;

  char* ws = (char*)d_ws;
  size_t o = 0;
  auto alloc = [&](size_t bytes) -> char* {
    char* p = ws + o; o += (bytes + 255) & ~(size_t)255; return p;
  };
  float*     latraw1 = (float*)alloc((size_t)H * 68 * 4);
  float*     latraw2 = (float*)alloc((size_t)H * 68 * 4);
  float*     feat1   = (float*)alloc((size_t)N * 64 * 4);
  float*     feat2   = (float*)alloc((size_t)N * 64 * 4);
  _Float16*  blat1   = (_Float16*)alloc((size_t)H * 64 * 2);
  _Float16*  blat2   = (_Float16*)alloc((size_t)H * 64 * 2);
  _Float16*  cat1    = (_Float16*)alloc((size_t)H * 128 * 2);
  _Float16*  lat2h   = (_Float16*)alloc((size_t)H * 64 * 2);
  _Float16*  aggb    = (_Float16*)alloc((size_t)H * 32 * 2);
  _Float16*  dtmp    = (_Float16*)alloc((size_t)H * 64 * 2);
  _Float16*  Asw1    = (_Float16*)alloc((size_t)64 * 1088 * 2);
  _Float16*  Apw1    = (_Float16*)alloc((size_t)128 * 1920 * 2);
  _Float16*  sw2p    = (_Float16*)alloc(64 * 64 * 2);
  _Float16*  cw1p    = (_Float16*)alloc(32 * 64 * 2);
  _Float16*  cw2p    = (_Float16*)alloc(32 * 32 * 2);
  _Float16*  dw1p    = (_Float16*)alloc(64 * 32 * 2);
  _Float16*  dw2p    = (_Float16*)alloc(64 * 64 * 2);
  _Float16*  pw2p    = (_Float16*)alloc(128 * 128 * 2);
  _Float16*  fw2p    = (_Float16*)alloc(128 * 128 * 2);
  _Float16*  fw3p    = (_Float16*)alloc(64 * 128 * 2);
  // aliased late-stage buffers (earlier buffers dead by then)
  _Float16*  pbuf = (_Float16*)latraw1;  // (H,128) f16 <= H*68*4 bytes
  _Float16*  p2   = (_Float16*)latraw2;  // (H,128)
  _Float16*  pts  = (_Float16*)feat1;    // (N,128)
  _Float16*  h2   = (_Float16*)feat2;    // (N,128)
  _Float16*  h3   = blat1;               // (N,64)

  (void)hipMemsetAsync(latraw1, 0, (size_t)H * 68 * 4, stream);
  (void)hipMemsetAsync(latraw2, 0, (size_t)H * 68 * 4, stream);

  k_pack_blur<<<(64 * 1088 + 255) / 256, 256, 0, stream>>>(sw1, Asw1, 64, 68, 72, 15, 1088);
  k_pack_blur<<<(128 * 1920 + 255) / 256, 256, 0, stream>>>(pw1, Apw1, 128, 128, 128, 15, 1920);
  k_pack_dense<<<16, 256, 0, stream>>>(sw2, sw2p, 64 * 64);
  k_pack_dense<<<8, 256, 0, stream>>>(cw1, cw1p, 32 * 64);
  k_pack_dense<<<4, 256, 0, stream>>>(cw2, cw2p, 32 * 32);
  k_pack_dense<<<8, 256, 0, stream>>>(dw1, dw1p, 64 * 32);
  k_pack_dense<<<16, 256, 0, stream>>>(dw2, dw2p, 64 * 64);
  k_pack_dense<<<64, 256, 0, stream>>>(pw2, pw2p, 128 * 128);
  k_pack_dense<<<64, 256, 0, stream>>>(fw2, fw2p, 128 * 128);
  k_pack_dense<<<32, 256, 0, stream>>>(fw3, fw3p, 64 * 128);

  k_mlp3<<<(N + 255) / 256, 256, 0, stream>>>(pc1, w1, b1, w2, b2, w3, b3, feat1, N);
  k_mlp3<<<(N + 255) / 256, 256, 0, stream>>>(pc2, w1, b1, w2, b2, w3, b3, feat2, N);

  k_splat<<<(N * 4 + 255) / 256, 256, 0, stream>>>(el1, feat1, bary1, off1, latraw1, N);
  k_splat<<<(N * 4 + 255) / 256, 256, 0, stream>>>(el2, feat2, bary2, off2, latraw2, N);

  k_blur_sw1<<<H / 32, 256, 0, stream>>>(latraw1, nb1, Asw1, sb1, blat1, H);
  k_blur_sw1<<<H / 32, 256, 0, stream>>>(latraw2, nb2, Asw1, sb1, blat2, H);

  // lat1 = lrelu(sw2 @ blat1 + sb2) -> cat1 cols 0..63 ; lat2 -> (H,64)
  k_gemm<<<(4 * (H / 32)) / 8, 256, 0, stream>>>(blat1, 64, sw2p, 64, 64, sb2, cat1, 128, 0, H, 1);
  k_gemm<<<(4 * (H / 32)) / 8, 256, 0, stream>>>(blat2, 64, sw2p, 64, 64, sb2, lat2h, 64, 0, H, 1);

  k_corr<<<H / 16, 64, 0, stream>>>(cat1, lat2h, cidx, cw1p, cb1, cw2p, cb2, aggb, H);

  k_gemm<<<(4 * (H / 32)) / 8, 256, 0, stream>>>(aggb, 32, dw1p, 32, 64, db1, dtmp, 64, 0, H, 1);
  k_gemm<<<(4 * (H / 32)) / 8, 256, 0, stream>>>(dtmp, 64, dw2p, 64, 64, db2, cat1, 128, 64, H, 1);

  k_pblur<<<H / 32, 256, 0, stream>>>(cat1, nb1, Apw1, pb1, pbuf, H);
  k_gemm<<<(8 * (H / 32)) / 8, 256, 0, stream>>>(pbuf, 128, pw2p, 128, 128, pb2, p2, 128, 0, H, 1);

  k_slice<<<(N * 4 + 255) / 256, 256, 0, stream>>>(p2, bary1, off1, pts, N);

  k_gemm<<<(8 * (N / 32)) / 8, 256, 0, stream>>>(pts, 128, fw2p, 128, 128, fb2, h2, 128, 0, N, 1);
  k_gemm<<<(4 * (N / 32)) / 8, 256, 0, stream>>>(h2, 128, fw3p, 128, 64, fb3, h3, 64, 0, N, 1);
  k_final<<<(N + 255) / 256, 256, 0, stream>>>(h3, fw4, fb4, (float*)d_out, N);
}
